// LabelSNN_62989990363219
// MI455X (gfx1250) — matrix operand order repaired and checked by reading, not compiled, not run on hardware
//
#include <hip/hip_runtime.h>

typedef __attribute__((ext_vector_type(16))) _Float16 v16h;
typedef __attribute__((ext_vector_type(8)))  float    v8f;

#define T_STEPS 128
#define BATCH   512
#define NOUT    1024
#define BM      64
#define BN      64
#define KC      128
#define DECAYF  0.95f
#define THRESHF 1.0f

// async global -> LDS copy, 16 bytes per lane (GVS mode: saddr base + 32-bit voffset)
// NOTE: the offset immediate is added to BOTH the global and LDS addresses (ISA 8.4.4),
// which matches our congruent 16B-step tile layout.
#define ASYNC_B128_0(lds, voff, base)                                          \
  asm volatile("global_load_async_to_lds_b128 %0, %1, %2"                      \
               :: "v"(lds), "v"(voff), "s"(base) : "memory")
#define ASYNC_B128(lds, voff, base, OFF)                                       \
  asm volatile("global_load_async_to_lds_b128 %0, %1, %2 offset:" #OFF        \
               :: "v"(lds), "v"(voff), "s"(base) : "memory")
#define WAIT_ASYNC0() asm volatile("s_wait_asynccnt 0x0" ::: "memory")

// ---------------- one-pass f32 -> f16 convert (spikes + weights) ----------------
__global__ __launch_bounds__(256) void cvt_f32_f16_kernel(
    const float* __restrict__ src, _Float16* __restrict__ dst, int n4)
{
  int i = blockIdx.x * blockDim.x + threadIdx.x;
  if (i < n4) {
    float4 f = ((const float4*)src)[i];
    union { _Float16 h[4]; uint64_t u; } p;
    p.h[0] = (_Float16)f.x; p.h[1] = (_Float16)f.y;
    p.h[2] = (_Float16)f.z; p.h[3] = (_Float16)f.w;
    ((uint64_t*)dst)[i] = p.u;
  }
}

// ---------------- fused GEMM (A @ W^T) + LIF recurrence ----------------
// A: f16 [T][BATCH][K], W: f16 [NOUT][K]. Membrane state lives in WMMA
// accumulators across all T steps; C is seeded with decay*v each step.
template<int K, bool STORE_SPK>
__global__ __launch_bounds__(256) void snn_layer_kernel(
    const _Float16* __restrict__ Aseq,
    const _Float16* __restrict__ Wh,
    _Float16*       __restrict__ spkOut,   // [T][BATCH][NOUT] (layer 1 only)
    float*          __restrict__ meanOut)  // [BATCH][NOUT]
{
  __shared__ __align__(16) _Float16 As[2][BM][KC];   // 2 x 16 KB
  __shared__ __align__(16) _Float16 Bs[2][BN][KC];   // 2 x 16 KB

  constexpr int KPC    = K / KC;          // chunks per timestep
  constexpr int NCHUNK = T_STEPS * KPC;

  const int tid  = threadIdx.x;
  const int lane = tid & 31;
  const int wave = tid >> 5;      // 0..7
  const int wm   = wave & 3;      // M tile (16 rows)
  const int wn   = wave >> 2;     // N pair (2 x 16 cols)

  const int blockM = blockIdx.x * BM;
  const int blockN = blockIdx.y * BN;

  // WMMA fragment lane decomposition (CDNA5 ISA 7.12.2)
  const int lm   = lane & 15;
  const int ka8  = (lane >> 4) * 8;
  const int kb16 = (lane >> 4) * 16;

  // async loader: 4 threads per row, 64 contiguous bytes (4 x b128) each
  const int ldRow = tid >> 2;
  const int ldCol = (tid & 3) * 32;                       // f16 elements
  const unsigned voff = (unsigned)((ldRow * K + ldCol) * 2);  // bytes, same for A and B
  const unsigned laA[2] = { (unsigned)(uintptr_t)&As[0][ldRow][ldCol],
                            (unsigned)(uintptr_t)&As[1][ldRow][ldCol] };
  const unsigned laB[2] = { (unsigned)(uintptr_t)&Bs[0][ldRow][ldCol],
                            (unsigned)(uintptr_t)&Bs[1][ldRow][ldCol] };

  auto issue = [&](int c) {
    const int tt  = c / KPC;
    const int kc  = c - tt * KPC;
    const int buf = c & 1;
    const _Float16* baseA = Aseq + ((size_t)tt * BATCH + blockM) * K + kc * KC;
    const _Float16* baseB = Wh + (size_t)blockN * K + kc * KC;
    const unsigned la = laA[buf], lb = laB[buf];
    ASYNC_B128_0(la, voff, baseA);
    ASYNC_B128(la, voff, baseA, 16);
    ASYNC_B128(la, voff, baseA, 32);
    ASYNC_B128(la, voff, baseA, 48);
    ASYNC_B128_0(lb, voff, baseB);
    ASYNC_B128(lb, voff, baseB, 16);
    ASYNC_B128(lb, voff, baseB, 32);
    ASYNC_B128(lb, voff, baseB, 48);
  };

  v8f v0 = {}, v1 = {};   // membrane potentials (two 16x16 C tiles)
  v8f s0 = {}, s1 = {};   // spike-count accumulators

  issue(0);   // prologue: stage chunk 0

  for (int t = 0; t < T_STEPS; ++t) {
    v8f c0 = v0 * DECAYF;   // v_pre = decay*v + sum_k ...
    v8f c1 = v1 * DECAYF;

    for (int kc = 0; kc < KPC; ++kc) {
      const int c   = t * KPC + kc;
      const int buf = c & 1;
      WAIT_ASYNC0();        // own async copies for chunk c landed in LDS
      __syncthreads();      // everyone's copies landed; prev compute done
      if (c + 1 < NCHUNK) issue(c + 1);   // prefetch next chunk into other buffer

      #pragma unroll
      for (int ks = 0; ks < KC / 32; ++ks) {
        const int k0 = ks * 32;
        v16h a, b0, b1;
        {
          float4* ap = (float4*)&a;
          ap[0] = *(const float4*)&As[buf][16 * wm + lm][k0 + kb16];
          ap[1] = *(const float4*)&As[buf][16 * wm + lm][k0 + kb16 + 8];
        }
        {
          float4* bp = (float4*)&b0;
          bp[0] = *(const float4*)&Bs[buf][32 * wn + lm][k0 + kb16];
          bp[1] = *(const float4*)&Bs[buf][32 * wn + lm][k0 + kb16 + 8];
        }
        {
          float4* bp = (float4*)&b1;
          bp[0] = *(const float4*)&Bs[buf][32 * wn + 16 + lm][k0 + kb16];
          bp[1] = *(const float4*)&Bs[buf][32 * wn + 16 + lm][k0 + kb16 + 8];
        }
        c0 = __builtin_amdgcn_wmma_f32_16x16x32_f16(false, a, false, b0,
                                                    (short)0, c0, false, false);
        c1 = __builtin_amdgcn_wmma_f32_16x16x32_f16(false, a, false, b1,
                                                    (short)0, c1, false, false);
      }
    }

    // LIF epilogue: spike, hard reset, accumulate (registers + global only)
    const int rh = (lane >> 4) * 8;
    #pragma unroll
    for (int e = 0; e < 8; ++e) {
      const int brow = blockM + 16 * wm + rh + e;
      {
        float vv = c0[e];
        float sp = vv > THRESHF ? 1.0f : 0.0f;
        s0[e] += sp;
        v0[e] = vv * (1.0f - sp);
        if constexpr (STORE_SPK)
          spkOut[((size_t)t * BATCH + brow) * NOUT + blockN + 32 * wn + lm] =
              (_Float16)sp;
      }
      {
        float vv = c1[e];
        float sp = vv > THRESHF ? 1.0f : 0.0f;
        s1[e] += sp;
        v1[e] = vv * (1.0f - sp);
        if constexpr (STORE_SPK)
          spkOut[((size_t)t * BATCH + brow) * NOUT + blockN + 32 * wn + 16 + lm] =
              (_Float16)sp;
      }
    }
  }

  // time-averaged spike rates
  const int rh = (lane >> 4) * 8;
  #pragma unroll
  for (int e = 0; e < 8; ++e) {
    const int brow = blockM + 16 * wm + rh + e;
    meanOut[(size_t)brow * NOUT + blockN + 32 * wn + lm]      = s0[e] * (1.0f / T_STEPS);
    meanOut[(size_t)brow * NOUT + blockN + 32 * wn + 16 + lm] = s1[e] * (1.0f / T_STEPS);
  }
}

extern "C" void kernel_launch(void* const* d_in, const int* in_sizes, int n_in,
                              void* d_out, int out_size, void* d_ws, size_t ws_size,
                              hipStream_t stream) {
  (void)in_sizes; (void)n_in; (void)out_size; (void)ws_size;
  const float* spikes = (const float*)d_in[0];   // [128][512][512] f32
  const float* W1     = (const float*)d_in[1];   // [1024][512]     f32
  const float* W2     = (const float*)d_in[2];   // [1024][1024]    f32
  float*       out    = (float*)d_out;

  // workspace carve-up (total ~195 MiB)
  char* w = (char*)d_ws;
  _Float16* spk1 = (_Float16*)w;                                   // 128 MiB
  _Float16* spkH = (_Float16*)(w + (size_t)134217728);             //  64 MiB
  _Float16* W1h  = (_Float16*)(w + (size_t)134217728 + 67108864);  //   1 MiB
  _Float16* W2h  = W1h + (size_t)NOUT * 512;                       //   2 MiB

  // f32 -> f16 pre-conversion (enables pure async-LDS staging in the GEMMs)
  {
    int n4 = T_STEPS * BATCH * 512 / 4;    // 8,388,608
    cvt_f32_f16_kernel<<<n4 / 256, 256, 0, stream>>>(spikes, spkH, n4);
    n4 = NOUT * 512 / 4;
    cvt_f32_f16_kernel<<<n4 / 256, 256, 0, stream>>>(W1, W1h, n4);
    n4 = NOUT * 1024 / 4;
    cvt_f32_f16_kernel<<<n4 / 256, 256, 0, stream>>>(W2, W2h, n4);
  }

  dim3 grid(BATCH / BM, NOUT / BN);   // (8, 16) = 128 workgroups
  dim3 block(256);                    // 8 wave32s

  // layer 1: spikes @ W1^T (K=512), emit spk1 + mean1
  snn_layer_kernel<512, true><<<grid, block, 0, stream>>>(spkH, W1h, spk1, out);
  // layer 2: spk1 @ W2^T (K=1024), emit mean2
  snn_layer_kernel<1024, false><<<grid, block, 0, stream>>>(
      spk1, W2h, nullptr, out + (size_t)BATCH * NOUT);
}
